// EdgeScoringHead_39539468926977
// MI455X (gfx1250) — matrix operand order repaired and checked by reading, not compiled, not run on hardware
//
#include <hip/hip_runtime.h>
#include <hip/hip_bf16.h>
#include <stdint.h>

typedef __attribute__((ext_vector_type(16))) __bf16 v16bf;
typedef __attribute__((ext_vector_type(8)))  float  v8f;

union V16 {
    uint4    q[2];
    uint32_t u[8];
    v16bf    v;
};

__device__ __forceinline__ uint16_t f2bf(float f) {
    union { float f; uint32_t u; } x; x.f = f;
    uint32_t r = x.u + 0x7FFFu + ((x.u >> 16) & 1u);   // round-to-nearest-even
    return (uint16_t)(r >> 16);
}
__device__ __forceinline__ uint32_t pack2(float lo, float hi) {
    return (uint32_t)f2bf(lo) | ((uint32_t)f2bf(hi) << 16);
}

// Block: 256 threads = 8 waves; wave w -> nt = w&3 (16-col tile), et = w>>2.
// Chunk = 64 edges; each wave computes 2 edge tiles (et*16 and et*16+32)
// against register-resident W1 fragments (12 x v16bf, ~96 VGPRs).
__global__ __launch_bounds__(256) void edge_score_wmma(
    const float* __restrict__ node, const long long* __restrict__ eidx,
    const float* __restrict__ W1, const float* __restrict__ b1,
    const float* __restrict__ W2, const float* __restrict__ b2,
    float* __restrict__ out, int E, int nChunks)
{
    // bf16 pair features, row stride 392 halves (784B -> bank stagger 4/row)
    __shared__ __align__(16) unsigned short feat[64][392];
    // per-nt, per-quad sub-partials (padded to 65 to avoid bank conflicts)
    __shared__ float psum[4][4][65];

    const int tid  = threadIdx.x;
    const int lane = tid & 31;
    const int w    = tid >> 5;
    const int nt   = w & 3;
    const int et   = w >> 2;
    const int n    = lane & 15;
    const int half = lane >> 4;
    const int col  = nt * 16 + n;

    // ---- W1 B-fragments -> registers (bf16), once per block ----
    // B (32x16): VGPR v: lanes 0-15 K=2v,2v+1 ; lanes 16-31 K=16+2v,16+2v+1
    V16 breg[12];
#pragma unroll
    for (int kk = 0; kk < 12; ++kk) {
#pragma unroll
        for (int v = 0; v < 8; ++v) {
            int k = kk * 32 + half * 16 + 2 * v;
            breg[kk].u[v] = pack2(W1[k * 64 + col], W1[(k + 1) * 64 + col]);
        }
    }
    const float b1v = b1[col];
    const float w2v = W2[col];
    const float b2v = b2[0];

    for (int chunk = blockIdx.x; chunk < nChunks; chunk += gridDim.x) {
        const int base = chunk * 64;

        // ---- Stage pair features for 64 edges into LDS (bf16) ----
        {
            const int e = tid >> 2;      // 0..63
            const int p = tid & 3;       // 0..3
            const long long s = eidx[base + e];
            const long long d = eidx[(long long)E + base + e];
            const float4* srow = reinterpret_cast<const float4*>(node + s * 128);
            const float4* drow = reinterpret_cast<const float4*>(node + d * 128);
#pragma unroll
            for (int i = 0; i < 8; ++i) {
                const int j = p + 4 * i;          // float4 index within row
                const float4 sv = srow[j];
                const float4 dv = drow[j];
                const int k = j * 4;
                uint32_t* f0 = reinterpret_cast<uint32_t*>(&feat[e][k]);
                f0[0] = pack2(sv.x, sv.y);
                f0[1] = pack2(sv.z, sv.w);
                uint32_t* f1 = reinterpret_cast<uint32_t*>(&feat[e][128 + k]);
                f1[0] = pack2(dv.x, dv.y);
                f1[1] = pack2(dv.z, dv.w);
                uint32_t* f2 = reinterpret_cast<uint32_t*>(&feat[e][256 + k]);
                f2[0] = pack2(fabsf(sv.x - dv.x), fabsf(sv.y - dv.y));
                f2[1] = pack2(fabsf(sv.z - dv.z), fabsf(sv.w - dv.w));
            }
        }
        __syncthreads();

        // ---- WMMA: two 16-edge tiles, K=384 in 12 steps, B in registers ----
        // A (16x32 bf16): lanes 0-15: VGPR0-3 = K 0..7, VGPR4-7 = K 8..15
        //                 lanes16-31: VGPR0-3 = K 16..23, VGPR4-7 = K 24..31
        const int row = et * 16 + n;
        v8f c0 = {}, c1 = {};
#pragma unroll
        for (int kk = 0; kk < 12; ++kk) {
            V16 a0, a1;
            a0.q[0] = *reinterpret_cast<const uint4*>(&feat[row][kk * 32 + half * 16]);
            a0.q[1] = *reinterpret_cast<const uint4*>(&feat[row][kk * 32 + 8 + half * 16]);
            a1.q[0] = *reinterpret_cast<const uint4*>(&feat[row + 32][kk * 32 + half * 16]);
            a1.q[1] = *reinterpret_cast<const uint4*>(&feat[row + 32][kk * 32 + 8 + half * 16]);
            c0 = __builtin_amdgcn_wmma_f32_16x16x32_bf16(
                     false, a0.v, false, breg[kk].v, (short)0, c0, false, false);
            c1 = __builtin_amdgcn_wmma_f32_16x16x32_bf16(
                     false, a1.v, false, breg[kk].v, (short)0, c1, false, false);
        }

        // ---- Epilogue: relu(C+b1)*W2, 2-stage shuffle, store sub-partials ----
#pragma unroll
        for (int tt = 0; tt < 2; ++tt) {
            const v8f c  = tt ? c1 : c0;
            const int tb = et * 16 + tt * 32;
#pragma unroll
            for (int r = 0; r < 8; ++r) {
                float h = c[r] + b1v;
                h = h > 0.0f ? h : 0.0f;
                float part = h * w2v;
                part += __shfl_xor(part, 1);
                part += __shfl_xor(part, 2);
                if ((n & 3) == 0)
                    psum[nt][n >> 2][tb + half * 8 + r] = part;
            }
        }
        __syncthreads();

        // ---- Finalize: fixed-order sum of 16 sub-partials, sigmoid, store ----
        if (tid < 64) {
            float acc = b2v;
#pragma unroll
            for (int ntI = 0; ntI < 4; ++ntI)
#pragma unroll
                for (int q = 0; q < 4; ++q)
                    acc += psum[ntI][q][tid];
            out[base + tid] = 1.0f / (1.0f + __expf(-acc));
        }
        __syncthreads();
    }
}

// Parallel tail for E % 64 != 0: one block per edge, 64 threads (one per
// hidden unit), fixed-order reduction -> deterministic, exact fp32.
__global__ __launch_bounds__(64) void edge_score_tail(
    const float* __restrict__ node, const long long* __restrict__ eidx,
    const float* __restrict__ W1, const float* __restrict__ b1,
    const float* __restrict__ W2, const float* __restrict__ b2,
    float* __restrict__ out, int E, int E0)
{
    __shared__ float hs[64];
    const int e = E0 + blockIdx.x;
    if (e >= E) return;
    const long long s = eidx[e];
    const long long d = eidx[(long long)E + e];
    const float* sh = node + s * 128;
    const float* dh = node + d * 128;
    const int nn = threadIdx.x;
    float hv = b1[nn];
    for (int k = 0; k < 128; ++k) {
        float sv = sh[k], dv = dh[k];
        hv += sv * W1[k * 64 + nn];
        hv += dv * W1[(128 + k) * 64 + nn];
        hv += fabsf(sv - dv) * W1[(256 + k) * 64 + nn];
    }
    hv = hv > 0.0f ? hv : 0.0f;
    hs[nn] = hv * W2[nn];
    __syncthreads();
    if (nn == 0) {
        float acc = b2[0];
        for (int i = 0; i < 64; ++i) acc += hs[i];
        out[e] = 1.0f / (1.0f + __expf(-acc));
    }
}

extern "C" void kernel_launch(void* const* d_in, const int* in_sizes, int n_in,
                              void* d_out, int out_size, void* d_ws, size_t ws_size,
                              hipStream_t stream) {
    const float*     node = (const float*)d_in[0];
    const long long* eidx = (const long long*)d_in[1];
    const float*     W1   = (const float*)d_in[2];
    const float*     b1   = (const float*)d_in[3];
    const float*     W2   = (const float*)d_in[4];
    const float*     b2   = (const float*)d_in[5];
    float* out = (float*)d_out;

    const int E = in_sizes[1] / 2;
    const int nChunks = E / 64;
    const int tail0 = nChunks * 64;

    if (nChunks > 0) {
        int blocks = nChunks < 2048 ? nChunks : 2048;
        edge_score_wmma<<<blocks, 256, 0, stream>>>(node, eidx, W1, b1, W2, b2,
                                                    out, E, nChunks);
    }
    if (tail0 < E) {
        int rem = E - tail0;
        edge_score_tail<<<rem, 64, 0, stream>>>(node, eidx, W1, b1, W2, b2,
                                                out, E, tail0);
    }
}
